// GNN_node_54374285967979
// MI455X (gfx1250) — compile-verified, matrix operand order checked by reading, not run on hardware
//
#include <hip/hip_runtime.h>
#include <math.h>

// ---------------- constants ----------------
#define NN 65536
#define EE 262144
#define DD 256
#define NF 9
#define EF 3
#define AV 128
#define BV 8
#define CH 20
#define WH 64
#define FN 3

typedef __attribute__((ext_vector_type(16))) __bf16 v16bf;
typedef __attribute__((ext_vector_type(8)))  float  v8f;

__device__ __forceinline__ float sigm(float x) { return 1.0f / (1.0f + expf(-x)); }

__device__ __forceinline__ unsigned long long makekey(float f, int n) {
  unsigned u = __float_as_uint(f);
  u = (u & 0x80000000u) ? ~u : (u | 0x80000000u);
  return ((unsigned long long)u << 32) | (unsigned)(~n);
}

// ---------------- utility ----------------
__global__ void k_zero(float* __restrict__ p, int n) {
  int i = blockIdx.x * 256 + threadIdx.x;
  if (i < n) p[i] = 0.0f;
}

// ---------------- atom embedding: x[n,c] = sum_f atom_emb[f, idx[n,f], c] ----------------
__global__ void k_atom_embed(const int* __restrict__ x_atom, const float* __restrict__ emb,
                             float* __restrict__ x) {
  int n = blockIdx.x, c = threadIdx.x;
  float acc = 0.f;
#pragma unroll
  for (int f = 0; f < NF; ++f) {
    int idx = x_atom[n * NF + f];
    acc += emb[((size_t)(f * AV + idx)) * DD + c];
  }
  x[(size_t)n * DD + c] = acc;
}

// ---------------- node MLPs: kappa + f0,f1,f2 (4 MLPs of D->CH->1) ----------------
__global__ void k_node_mlps(const float* __restrict__ x,
                            const float* __restrict__ cw1, const float* __restrict__ cb1,
                            const float* __restrict__ cw2, const float* __restrict__ cb2,
                            const float* __restrict__ fw1, const float* __restrict__ fb1,
                            const float* __restrict__ fw2, const float* __restrict__ fb2,
                            float* __restrict__ kappa, float* __restrict__ fout) {
  __shared__ float xs[DD];
  __shared__ float hh[4 * CH];
  int n = blockIdx.x, tid = threadIdx.x;
  for (int i = tid; i < DD; i += 128) xs[i] = x[(size_t)n * DD + i];
  __syncthreads();
  if (tid < 4 * CH) {
    int which = tid / CH, j = tid % CH;
    const float* w1 = (which == 0) ? cw1 : fw1 + (size_t)(which - 1) * DD * CH;
    const float* b1 = (which == 0) ? cb1 : fb1 + (which - 1) * CH;
    float acc = b1[j];
    for (int k = 0; k < DD; ++k) acc += xs[k] * w1[k * CH + j];
    hh[tid] = fmaxf(acc, 0.f);
  }
  __syncthreads();
  if (tid < 4) {
    const float* w2 = (tid == 0) ? cw2 : fw2 + (tid - 1) * CH;
    float b2 = (tid == 0) ? cb2[0] : fb2[tid - 1];
    float acc = b2;
#pragma unroll
    for (int j = 0; j < CH; ++j) acc += hh[tid * CH + j] * w2[j];
    float o = sigm(acc);
    if (tid == 0) kappa[n] = o;
    else fout[(size_t)(tid - 1) * NN + n] = o;
  }
}

// ---------------- generic sum reduce -> atomic ----------------
__global__ void k_reduce_sum(const float* __restrict__ v, int n, float* __restrict__ out) {
  __shared__ float sd[256];
  int tid = threadIdx.x, i = blockIdx.x * 256 + tid;
  sd[tid] = (i < n) ? v[i] : 0.f;
  __syncthreads();
  for (int s = 128; s > 0; s >>= 1) { if (tid < s) sd[tid] += sd[tid + s]; __syncthreads(); }
  if (tid == 0) atomicAdd(out, sd[0]);
}

// ---------------- WeightMLP ----------------
// stage 1: out64[j] = sum_e s_e * wm_w1[e, j]
__global__ void k_wm1(const int* __restrict__ ea, const float* __restrict__ w1,
                      float* __restrict__ out64) {
  __shared__ float red[4][WH];
  int tid = threadIdx.x, j = tid & 63, g = tid >> 6;
  int base = blockIdx.x * 1024;
  float acc = 0.f;
  for (int e = base + g; e < base + 1024; e += 4) {
    float s = (float)(ea[e * 3] + ea[e * 3 + 1] + ea[e * 3 + 2]);
    acc += s * w1[(size_t)e * WH + j];
  }
  red[g][j] = acc;
  __syncthreads();
  if (g == 0) atomicAdd(&out64[j], red[0][j] + red[1][j] + red[2][j] + red[3][j]);
}
// stage 2: h1 = relu(relu(out64+b1) @ w2 + b2)   (single block, 64 threads)
__global__ void k_wm2(const float* __restrict__ out64, const float* __restrict__ b1,
                      const float* __restrict__ w2, const float* __restrict__ b2,
                      float* __restrict__ h1) {
  __shared__ float h0[WH];
  int j = threadIdx.x;
  h0[j] = fmaxf(out64[j] + b1[j], 0.f);
  __syncthreads();
  float acc = b2[j];
#pragma unroll 8
  for (int k = 0; k < WH; ++k) acc += h0[k] * w2[k * WH + j];
  h1[j] = fmaxf(acc, 0.f);
}
// stage 3: weights[e] = sigmoid(sum_j h1[j]*w3[j,e] + b3[e])
__global__ void k_weights(const float* __restrict__ h1, const float* __restrict__ w3,
                          const float* __restrict__ b3, float* __restrict__ w) {
  __shared__ float h1s[WH];
  int tid = threadIdx.x;
  if (tid < WH) h1s[tid] = h1[tid];
  __syncthreads();
  int e = blockIdx.x * 256 + tid;
  float acc = b3[e];
#pragma unroll 8
  for (int j = 0; j < WH; ++j) acc += h1s[j] * w3[(size_t)j * EE + e];
  w[e] = sigm(acc);
}

// ---------------- curvature-loss edge passes ----------------
__global__ void k_fn_pass1(const float* __restrict__ f, const float* __restrict__ w,
                           const int* __restrict__ src, const int* __restrict__ dst,
                           float* __restrict__ gamma, float* __restrict__ dlf) {
  int e = blockIdx.x * 256 + threadIdx.x;
  int s = src[e], d = dst[e];
  float fd = f[d] - f[s];
  float we = w[e];
  atomicAdd(&gamma[s], 0.5f * we * fd * fd);
  atomicAdd(&dlf[s], we * fd);
}
__global__ void k_fn_pass2(const float* __restrict__ f, const float* __restrict__ w,
                           const int* __restrict__ src, const int* __restrict__ dst,
                           const float* __restrict__ gamma, const float* __restrict__ dlf,
                           float* __restrict__ dgm, float* __restrict__ gfd) {
  int e = blockIdx.x * 256 + threadIdx.x;
  int s = src[e], d = dst[e];
  float we = w[e];
  float fd = f[d] - f[s];
  float gd = gamma[d] - gamma[s];
  float dfd = dlf[d] - dlf[s];
  atomicAdd(&dgm[s], we * gd);
  atomicAdd(&gfd[s], 0.5f * we * fd * dfd);
}
__global__ void k_fn_reduce(const float* __restrict__ kappa, const float* __restrict__ gamma,
                            const float* __restrict__ dgm, const float* __restrict__ gfd,
                            float* __restrict__ loss) {
  __shared__ float sd[256];
  int tid = threadIdx.x, i = blockIdx.x * 256 + tid;
  float gamma2 = 0.5f * dgm[i] - gfd[i];
  sd[tid] = fmaxf(kappa[i] * gamma[i] - gamma2, 0.f);
  __syncthreads();
  for (int s = 128; s > 0; s >>= 1) { if (tid < s) sd[tid] += sd[tid + s]; __syncthreads(); }
  if (tid == 0) atomicAdd(loss, sd[0]);
}

// ---------------- deterministic top-k threshold via 64-bit-key binary search ----------------
__global__ void k_topk(const float* __restrict__ kappa, const int* __restrict__ p_ptr,
                       unsigned long long* __restrict__ thr_out) {
  __shared__ int cnt[1024];
  int tid = threadIdx.x;
  long long nr = (long long)NN * p_ptr[0] / 100;
  int K = (nr < 100) ? (int)nr : 100;
  if (K <= 0) { if (tid == 0) *thr_out = 0xFFFFFFFFFFFFFFFFULL; return; }
  unsigned long long lo = 0, hi = 0xFFFFFFFFFFFFFFFFULL;
  for (int it = 0; it < 64; ++it) {
    if (lo >= hi) break;
    unsigned long long mid = lo + ((hi - lo + 1) >> 1);
    int c = 0;
    for (int i = tid; i < NN; i += 1024)
      if (makekey(kappa[i], i) >= mid) c++;
    cnt[tid] = c;
    __syncthreads();
    for (int s = 512; s > 0; s >>= 1) { if (tid < s) cnt[tid] += cnt[tid + s]; __syncthreads(); }
    int total = cnt[0];
    __syncthreads();
    if (total >= K) lo = mid; else hi = mid - 1;
  }
  if (tid == 0) *thr_out = lo;
}
__global__ void k_ew(const float* __restrict__ kappa, const int* __restrict__ src,
                     const int* __restrict__ dst, const unsigned long long* __restrict__ thr,
                     float* __restrict__ ew) {
  int e = blockIdx.x * 256 + threadIdx.x;
  unsigned long long T = *thr;
  int s = src[e], d = dst[e];
  bool pruned = (makekey(kappa[s], s) >= T) || (makekey(kappa[d], d) >= T);
  ew[e] = pruned ? 1e-5f : 1.0f;
}

// ---------------- GIN: z0 = (1+eps)*h ; then edge atomic accumulate ----------------
__global__ void k_scale_init(const float* __restrict__ h, const float* __restrict__ eps,
                             float* __restrict__ z0, int n) {
  int i = blockIdx.x * 256 + threadIdx.x;
  if (i < n) z0[i] = (1.0f + eps[0]) * h[i];
}
__global__ void k_edge_gin(const float* __restrict__ h, const float* __restrict__ be,
                           const int* __restrict__ ea, const int* __restrict__ src,
                           const int* __restrict__ dst, const float* __restrict__ ew,
                           float* __restrict__ z0) {
  int e = blockIdx.x, c = threadIdx.x;
  int s = src[e], d = dst[e];
  int a0 = ea[e * 3], a1 = ea[e * 3 + 1], a2 = ea[e * 3 + 2];
  float bond = be[(size_t)(0 * BV + a0) * DD + c] +
               be[(size_t)(1 * BV + a1) * DD + c] +
               be[(size_t)(2 * BV + a2) * DD + c];
  float m = ew[e] * fmaxf(h[(size_t)s * DD + c] + bond, 0.f);
  atomicAdd(&z0[(size_t)d * DD + c], m);
}

// ---------------- weight transpose + bf16 convert: Wt[c][k] = W[k][c] ----------------
__global__ void k_convT(const float* __restrict__ W, __bf16* __restrict__ Wt, int K, int C) {
  int i = blockIdx.x * 256 + threadIdx.x;
  if (i < K * C) {
    int k = i / C, c = i % C;
    Wt[(size_t)c * K + k] = (__bf16)W[i];
  }
}

// ---------------- WMMA GEMM: C[N,Cc] = A[N,K] @ B[K,Cc] + bias ----------------
// Bt is column-major B (i.e. Bt[c][k]) in bf16. 8 waves/block, 16 rows x 128 cols per block.
template <bool A_F32>
__global__ __launch_bounds__(256) void k_gemm(const void* __restrict__ Aptr,
                                              const __bf16* __restrict__ Bt,
                                              const float* __restrict__ bias,
                                              float* __restrict__ C, int K, int Cc) {
  __shared__ __bf16 Ash[16][512 + 8];
  const int tid = threadIdx.x;
  const int wave = tid >> 5;
  const int lane = tid & 31;
  const int ntc = Cc >> 7;
  const int rowblk = blockIdx.x / ntc;
  const int colblk = blockIdx.x % ntc;
  const int row0 = rowblk * 16;
  const int ccol = colblk * 128 + wave * 16 + (lane & 15);

  // stage A tile (16 x K) into LDS as bf16
  if (A_F32) {
    const float* A = (const float*)Aptr;
    for (int i = tid; i < 16 * K; i += 256) {
      int r = i / K, k = i - r * K;
      Ash[r][k] = (__bf16)A[(size_t)(row0 + r) * K + k];
    }
  } else {
    const __bf16* A = (const __bf16*)Aptr;
    for (int i = tid; i < 16 * K; i += 256) {
      int r = i / K, k = i - r * K;
      Ash[r][k] = A[(size_t)(row0 + r) * K + k];
    }
  }
  __syncthreads();

  const int mrow = lane & 15;
  const int kb = (lane < 16) ? 0 : 8;            // ISA 16-bit A/B fragment layout
  const __bf16* Bcol = Bt + (size_t)ccol * K;

  v8f acc = {};
  for (int k0 = 0; k0 < K; k0 += 32) {
    v16bf a, b;
#pragma unroll
    for (int i = 0; i < 8; ++i) {
      a[i]     = Ash[mrow][k0 + kb + i];
      a[8 + i] = Ash[mrow][k0 + 16 + kb + i];
      b[i]     = Bcol[k0 + kb + i];
      b[8 + i] = Bcol[k0 + 16 + kb + i];
    }
    acc = __builtin_amdgcn_wmma_f32_16x16x32_bf16(false, a, false, b, (short)0, acc,
                                                  false, false);
  }
  const float bv = bias[ccol];
  const int mbase = (lane < 16) ? 0 : 8;
#pragma unroll
  for (int r = 0; r < 8; ++r)
    C[(size_t)(row0 + mbase + r) * Cc + ccol] = acc[r] + bv;
}

// ---------------- BN column stats ----------------
__global__ void k_colstats(const float* __restrict__ T, int C,
                           float* __restrict__ sum, float* __restrict__ sq) {
  int col = blockIdx.x * 256 + threadIdx.x;
  if (col >= C) return;
  int r0 = blockIdx.y * 1024;
  float s = 0.f, q = 0.f;
  for (int r = r0; r < r0 + 1024; ++r) {
    float v = T[(size_t)r * C + col];
    s += v; q += v * v;
  }
  atomicAdd(&sum[col], s);
  atomicAdd(&sq[col], q);
}
// BN + ReLU -> bf16 (feeds GEMM2)
__global__ void k_bn_relu_bf(const float* __restrict__ T, const float* __restrict__ sum,
                             const float* __restrict__ sq, const float* __restrict__ g,
                             const float* __restrict__ b, __bf16* __restrict__ O, int C) {
  int i = blockIdx.x * 256 + threadIdx.x;
  int col = i % C;
  float mean = sum[col] * (1.0f / NN);
  float var = sq[col] * (1.0f / NN) - mean * mean;
  float v = (T[i] - mean) * rsqrtf(var + 1e-5f) * g[col] + b[col];
  O[i] = (__bf16)fmaxf(v, 0.f);
}
// BN (+optional ReLU) -> f32
__global__ void k_bn_out(const float* __restrict__ T, const float* __restrict__ sum,
                         const float* __restrict__ sq, const float* __restrict__ g,
                         const float* __restrict__ b, float* __restrict__ O, int C, int relu) {
  int i = blockIdx.x * 256 + threadIdx.x;
  int col = i % C;
  float mean = sum[col] * (1.0f / NN);
  float var = sq[col] * (1.0f / NN) - mean * mean;
  float v = (T[i] - mean) * rsqrtf(var + 1e-5f) * g[col] + b[col];
  O[i] = relu ? fmaxf(v, 0.f) : v;
}

// ---------------- finalize scalar ----------------
__global__ void k_final(const float* __restrict__ loss, const float* __restrict__ kapsum,
                        float* __restrict__ out) {
  out[0] = loss[0] - 3.0f * kapsum[0];
}

// ---------------- host ----------------
extern "C" void kernel_launch(void* const* d_in, const int* in_sizes, int n_in,
                              void* d_out, int out_size, void* d_ws, size_t ws_size,
                              hipStream_t stream) {
  (void)in_sizes; (void)n_in; (void)out_size; (void)ws_size;
  const int* x_atom   = (const int*)d_in[0];
  const int* ei       = (const int*)d_in[1];
  const int* ea       = (const int*)d_in[2];
  const int* p        = (const int*)d_in[3];
  const float* atom_e = (const float*)d_in[4];
  const float* bond_e = (const float*)d_in[5];
  const float* gw1    = (const float*)d_in[6];
  const float* gb1    = (const float*)d_in[7];
  const float* gbng   = (const float*)d_in[8];
  const float* gbnb   = (const float*)d_in[9];
  const float* gw2    = (const float*)d_in[10];
  const float* gb2    = (const float*)d_in[11];
  const float* geps   = (const float*)d_in[12];
  const float* bng    = (const float*)d_in[13];
  const float* bnb    = (const float*)d_in[14];
  const float* cw1    = (const float*)d_in[15];
  const float* cb1    = (const float*)d_in[16];
  const float* cw2    = (const float*)d_in[17];
  const float* cb2    = (const float*)d_in[18];
  const float* fw1    = (const float*)d_in[19];
  const float* fb1    = (const float*)d_in[20];
  const float* fw2    = (const float*)d_in[21];
  const float* fb2    = (const float*)d_in[22];
  const float* wmw1   = (const float*)d_in[23];
  const float* wmb1   = (const float*)d_in[24];
  const float* wmw2   = (const float*)d_in[25];
  const float* wmb2   = (const float*)d_in[26];
  const float* wmw3   = (const float*)d_in[27];
  const float* wmb3   = (const float*)d_in[28];
  float* out = (float*)d_out;

  const int* src = ei;
  const int* dst = ei + EE;

  char* ws = (char*)d_ws;
  size_t off = 0;
  auto take = [&](size_t bytes) -> char* {
    char* r = ws + off;
    off = (off + bytes + 255) & ~(size_t)255;
    return r;
  };
  float*  h      = (float*)take((size_t)NN * DD * 4);
  float*  z0     = (float*)take((size_t)NN * DD * 4);
  float*  t      = (float*)take((size_t)NN * 2 * DD * 4);
  __bf16* tbf    = (__bf16*)take((size_t)NN * 2 * DD * 2);
  float*  z      = (float*)take((size_t)NN * DD * 4);
  __bf16* w1t    = (__bf16*)take((size_t)DD * 2 * DD * 2);
  __bf16* w2t    = (__bf16*)take((size_t)2 * DD * DD * 2);
  float*  kappa  = (float*)take((size_t)NN * 4);
  float*  f3     = (float*)take((size_t)FN * NN * 4);
  float*  gamma4 = (float*)take((size_t)4 * NN * 4);  // gamma|dlf|dgm|gfd
  float*  wts    = (float*)take((size_t)EE * 4);
  float*  ew     = (float*)take((size_t)EE * 4);
  float*  wm0    = (float*)take(WH * 4);
  float*  wmh1   = (float*)take(WH * 4);
  float*  stats  = (float*)take(1024 * 4);            // sum[0..511] | sq[512..1023]
  float*  scal   = (float*)take(16 * 4);              // [0]=loss [1]=kapsum
  unsigned long long* thr = (unsigned long long*)take(8);

  float* gamma = gamma4, *dlf = gamma4 + NN, *dgm = gamma4 + 2 * NN, *gfd = gamma4 + 3 * NN;
  float* ssum = stats, *ssq = stats + 512;

  // zero accumulators (every call: graph-replay safe)
  k_zero<<<1, 256, 0, stream>>>(wm0, WH);
  k_zero<<<1, 256, 0, stream>>>(scal, 16);

  // node features + small MLPs
  k_atom_embed<<<NN, 256, 0, stream>>>(x_atom, atom_e, h);
  k_node_mlps<<<NN, 128, 0, stream>>>(h, cw1, cb1, cw2, cb2, fw1, fb1, fw2, fb2, kappa, f3);
  k_reduce_sum<<<NN / 256, 256, 0, stream>>>(kappa, NN, scal + 1);

  // WeightMLP -> edge weights
  k_wm1<<<EE / 1024, 256, 0, stream>>>(ea, wmw1, wm0);
  k_wm2<<<1, WH, 0, stream>>>(wm0, wmb1, wmw2, wmb2, wmh1);
  k_weights<<<EE / 256, 256, 0, stream>>>(wmh1, wmw3, wmb3, wts);

  // curvature loss (3 fns)
  for (int fn = 0; fn < FN; ++fn) {
    k_zero<<<(4 * NN) / 256, 256, 0, stream>>>(gamma4, 4 * NN);
    const float* f = f3 + (size_t)fn * NN;
    k_fn_pass1<<<EE / 256, 256, 0, stream>>>(f, wts, src, dst, gamma, dlf);
    k_fn_pass2<<<EE / 256, 256, 0, stream>>>(f, wts, src, dst, gamma, dlf, dgm, gfd);
    k_fn_reduce<<<NN / 256, 256, 0, stream>>>(kappa, gamma, dgm, gfd, scal);
  }

  // top-k prune -> ew
  k_topk<<<1, 1024, 0, stream>>>(kappa, p, thr);
  k_ew<<<EE / 256, 256, 0, stream>>>(kappa, src, dst, thr, ew);

  // GIN layers
  for (int l = 0; l < 3; ++l) {
    k_convT<<<(DD * 2 * DD) / 256, 256, 0, stream>>>(gw1 + (size_t)l * DD * 2 * DD, w1t, DD, 2 * DD);
    k_convT<<<(2 * DD * DD) / 256, 256, 0, stream>>>(gw2 + (size_t)l * 2 * DD * DD, w2t, 2 * DD, DD);
    k_scale_init<<<(NN * DD) / 256, 256, 0, stream>>>(h, geps + l, z0, NN * DD);
    k_edge_gin<<<EE, 256, 0, stream>>>(h, bond_e + (size_t)l * EF * BV * DD, ea, src, dst, ew, z0);

    // GEMM1: t = z0 @ w1 + b1  [N,256]x[256,512]
    k_gemm<true><<<(NN / 16) * (2 * DD / 128), 256, 0, stream>>>(z0, w1t, gb1 + (size_t)l * 2 * DD,
                                                                 t, DD, 2 * DD);
    k_zero<<<4, 256, 0, stream>>>(stats, 1024);
    k_colstats<<<dim3(2, NN / 1024), 256, 0, stream>>>(t, 2 * DD, ssum, ssq);
    k_bn_relu_bf<<<(NN * 2 * DD) / 256, 256, 0, stream>>>(t, ssum, ssq,
                                                          gbng + (size_t)l * 2 * DD,
                                                          gbnb + (size_t)l * 2 * DD, tbf, 2 * DD);

    // GEMM2: z = relu_bn(t) @ w2 + b2  [N,512]x[512,256]
    k_gemm<false><<<(NN / 16) * (DD / 128), 256, 0, stream>>>(tbf, w2t, gb2 + (size_t)l * DD,
                                                              z, 2 * DD, DD);
    k_zero<<<4, 256, 0, stream>>>(stats, 1024);
    k_colstats<<<dim3(1, NN / 1024), 256, 0, stream>>>(z, DD, ssum, ssq);
    float* outp = (l == 2) ? out : h;
    k_bn_out<<<(NN * DD) / 256, 256, 0, stream>>>(z, ssum, ssq, bng + (size_t)l * DD,
                                                  bnb + (size_t)l * DD, outp, DD, (l < 2) ? 1 : 0);
  }

  k_final<<<1, 1, 0, stream>>>(scal, scal + 1, out + (size_t)NN * DD);
}
